// LinearAttention_64141041598845
// MI455X (gfx1250) — matrix operand order repaired and checked by reading, not compiled, not run on hardware
//
#include <hip/hip_runtime.h>
#include <hip/hip_bf16.h>

typedef __attribute__((ext_vector_type(16))) __bf16 v16bf;
typedef __attribute__((ext_vector_type(8)))  float  v8f;

#define B_     4
#define S_     2048
#define HID_   2048
#define NH_    16
#define HD_    128
#define NQKV_  6144
#define M_     (B_ * S_)           // 8192
#define CHUNK_ 128
#define NC_    (S_ / CHUNK_)       // 16
#define OUT_ELEMS_ ((size_t)B_ * S_ * HID_)   // 16777216

#define HAS_TDM (__has_builtin(__builtin_amdgcn_tensor_load_to_lds) && \
                 __has_builtin(__builtin_amdgcn_s_wait_tensorcnt))

// ---------------- helpers ----------------

__device__ __forceinline__ unsigned short f2bf(float f) {
    unsigned u = __float_as_uint(f);
    unsigned r = u + 0x7FFFu + ((u >> 16) & 1u);  // round-to-nearest-even
    return (unsigned short)(r >> 16);
}

__device__ __forceinline__ float phi_elu1(float x) {
    return x > 0.f ? x + 1.f : __expf(x);   // elu(x)+1
}

__device__ __forceinline__ float sigmoidf(float x) {
    return 1.f / (1.f + __expf(-x));
}

union FragBF { v16bf v; uint4 u[2]; };

// ---------------- cast x -> bf16 ----------------

__global__ void cast_x_kernel(const float* __restrict__ x,
                              unsigned short* __restrict__ xb, int n) {
    int i = (blockIdx.x * blockDim.x + threadIdx.x) * 4;
    if (i >= n) return;
    float4 f = *(const float4*)(x + i);
    ushort4 o;
    o.x = f2bf(f.x); o.y = f2bf(f.y); o.z = f2bf(f.z); o.w = f2bf(f.w);
    *(ushort4*)(xb + i) = o;
}

// ---------------- transpose + cast W[K,N] -> Wt[N,K] bf16 ----------------

__global__ void transpose_cast_kernel(const float* __restrict__ W,
                                      unsigned short* __restrict__ Wt,
                                      int K, int N) {
    __shared__ unsigned short tile[32][33];
    int nb = blockIdx.x * 32, kb = blockIdx.y * 32;
    int tx = threadIdx.x, ty = threadIdx.y;   // blockDim = (32, 8)
#pragma unroll
    for (int r = 0; r < 4; ++r) {
        int k = kb + ty + r * 8;
        tile[ty + r * 8][tx] = f2bf(W[(size_t)k * N + nb + tx]);
    }
    __syncthreads();
#pragma unroll
    for (int r = 0; r < 4; ++r) {
        int n = nb + ty + r * 8;
        Wt[(size_t)n * K + kb + tx] = tile[tx][ty + r * 8];
    }
}

// ---------------- shared GEMM epilogue ----------------

__device__ __forceinline__ void gemm_epilogue(
    v8f acc[2][4], float* __restrict__ Out, const float* __restrict__ mask,
    int baseM, int baseN, int laneHi, int lane15, int N, int mode) {
#pragma unroll
    for (int i = 0; i < 2; ++i) {
#pragma unroll
        for (int j = 0; j < 4; ++j) {
            const int col = baseN + j * 16 + lane15;
#pragma unroll
            for (int r = 0; r < 8; ++r) {
                const int row = baseM + i * 16 + laneHi * 8 + r;
                float v = acc[i][j][r];
                if (mode == 1) {
                    int part = col >> 11;     // 0:q 1:k 2:v
                    if (part == 0) {
                        v = phi_elu1(v);
                    } else {
                        float m = mask[row];
                        v = (part == 1) ? phi_elu1(v) * m : v * m;
                    }
                }
                Out[(size_t)row * N + col] = v;
            }
        }
    }
}

#if HAS_TDM
// =============== TDM-staged bf16 WMMA GEMM ===============

typedef unsigned int v4u __attribute__((ext_vector_type(4)));
typedef int          v8i __attribute__((ext_vector_type(8)));
typedef int          v4i __attribute__((ext_vector_type(4)));

#define KC_        64
#define ROWB_      144                      // 128B row + 16B pad
#define A_BYTES_   (64  * ROWB_)            //  9216
#define B_BYTES_   (256 * ROWB_)            // 36864
#define BUF_BYTES_ (A_BYTES_ + B_BYTES_)    // 46080

__device__ __forceinline__ void tdm_load_2d(
    unsigned int lds_addr, const void* tile_gaddr,
    unsigned int tensor_d0, unsigned int tensor_d1, unsigned int stride_d0,
    unsigned int tile_d0, unsigned int tile_d1) {
    unsigned long long ga = (unsigned long long)(uintptr_t)tile_gaddr;
    v4u g0;
    g0[0] = 1u;                                        // count=1
    g0[1] = lds_addr;
    g0[2] = (unsigned int)ga;
    g0[3] = (unsigned int)(ga >> 32) | (2u << 30);     // type=2 ("image")
    v8i g1;
    g1[0] = (int)((1u << 16) | (1u << 20) | (4u << 22) | (3u << 25));
    g1[1] = (int)((tensor_d0 & 0xFFFFu) << 16);
    g1[2] = (int)(((tensor_d0 >> 16) & 0xFFFFu) | ((tensor_d1 & 0xFFFFu) << 16));
    g1[3] = (int)(((tensor_d1 >> 16) & 0xFFFFu) | (tile_d0 << 16));
    g1[4] = (int)tile_d1;
    g1[5] = (int)stride_d0;
    g1[6] = 0;
    g1[7] = 0;
    v4i z = {0, 0, 0, 0};
#if __clang_major__ >= 23
    v8i z8 = {0, 0, 0, 0, 0, 0, 0, 0};
    __builtin_amdgcn_tensor_load_to_lds(g0, g1, z, z, z8, 0);
#else
    __builtin_amdgcn_tensor_load_to_lds(g0, g1, z, z, 0);
#endif
}

__global__ __launch_bounds__(256) void wmma_gemm_tdm_kernel(
    const unsigned short* __restrict__ A,
    const unsigned short* __restrict__ Bt,
    float* __restrict__ Out,
    const float* __restrict__ mask,
    int N, int K, int mode)
{
    __shared__ __attribute__((aligned(64))) unsigned char smem[2 * BUF_BYTES_];

    const int lane   = threadIdx.x & 31;
    const int wave   = threadIdx.x >> 5;
    const int laneHi = lane >> 4;
    const int lane15 = lane & 15;
    const int waveM  = wave >> 2;
    const int waveN  = wave & 3;
    const int baseM  = blockIdx.y * 64  + waveM * 32;
    const int baseN  = blockIdx.x * 256 + waveN * 64;

    v8f acc[2][4];
#pragma unroll
    for (int i = 0; i < 2; ++i)
#pragma unroll
        for (int j = 0; j < 4; ++j)
#pragma unroll
            for (int r = 0; r < 8; ++r) acc[i][j][r] = 0.f;

    const unsigned int ldsBase = (unsigned int)(uintptr_t)(&smem[0]);
    const unsigned short* aG = A  + (size_t)(blockIdx.y * 64)  * K;
    const unsigned short* bG = Bt + (size_t)(blockIdx.x * 256) * K;
    const int NKC = K / KC_;

    if (wave == 0) {
        tdm_load_2d(ldsBase,            aG, K, M_, K, KC_, 64);
        tdm_load_2d(ldsBase + A_BYTES_, bG, K, N,  K, KC_, 256);
    }

    for (int kc = 0; kc < NKC; ++kc) {
        const unsigned char* buf = smem + (size_t)(kc & 1) * BUF_BYTES_;
        if (wave == 0) {
            if (kc + 1 < NKC) {
                unsigned int nb = ldsBase + (unsigned int)((kc + 1) & 1) * BUF_BYTES_;
                int k0n = (kc + 1) * KC_;
                tdm_load_2d(nb,            aG + k0n, K, M_, K, KC_, 64);
                tdm_load_2d(nb + A_BYTES_, bG + k0n, K, N,  K, KC_, 256);
                __builtin_amdgcn_s_wait_tensorcnt(2);
            } else {
                __builtin_amdgcn_s_wait_tensorcnt(0);
            }
        }
        __syncthreads();

        const unsigned char* aB = buf;
        const unsigned char* bB = buf + A_BYTES_;
#pragma unroll
        for (int sub = 0; sub < 2; ++sub) {
            FragBF a[2], bf[4];
#pragma unroll
            for (int i = 0; i < 2; ++i) {
                const unsigned char* p = aB +
                    (size_t)(waveM * 32 + i * 16 + lane15) * ROWB_ + sub * 64 + laneHi * 16;
                a[i].u[0] = *(const uint4*)(p);
                a[i].u[1] = *(const uint4*)(p + 32);
            }
#pragma unroll
            for (int j = 0; j < 4; ++j) {
                const unsigned char* p = bB +
                    (size_t)(waveN * 64 + j * 16 + lane15) * ROWB_ + sub * 64 + laneHi * 16;
                bf[j].u[0] = *(const uint4*)(p);
                bf[j].u[1] = *(const uint4*)(p + 32);
            }
#pragma unroll
            for (int j = 0; j < 4; ++j) {
                acc[0][j] = __builtin_amdgcn_wmma_f32_16x16x32_bf16(
                    false, a[0].v, false, bf[j].v, (short)0, acc[0][j], false, false);
                acc[1][j] = __builtin_amdgcn_wmma_f32_16x16x32_bf16(
                    false, a[1].v, false, bf[j].v, (short)0, acc[1][j], false, false);
            }
        }
        __syncthreads();
    }

    gemm_epilogue(acc, Out, mask, baseM, baseN, laneHi, lane15, N, mode);
}

#else  // !HAS_TDM fallback

__global__ __launch_bounds__(256) void wmma_gemm_kernel(
    const unsigned short* __restrict__ A,
    const unsigned short* __restrict__ Bt,
    float* __restrict__ Out,
    const float* __restrict__ mask,
    int N, int K, int mode)
{
    const int lane   = threadIdx.x & 31;
    const int wave   = threadIdx.x >> 5;
    const int laneHi = lane >> 4;
    const int lane15 = lane & 15;

    const int baseM = blockIdx.y * 64  + (wave >> 2) * 32;
    const int baseN = blockIdx.x * 256 + (wave & 3)  * 64;

    v8f acc[2][4];
#pragma unroll
    for (int i = 0; i < 2; ++i)
#pragma unroll
        for (int j = 0; j < 4; ++j)
#pragma unroll
            for (int r = 0; r < 8; ++r) acc[i][j][r] = 0.f;

    const unsigned short* aPtr0 = A + (size_t)(baseM + lane15) * K + laneHi * 8;
    const unsigned short* aPtr1 = aPtr0 + (size_t)16 * K;
    const unsigned short* bPtr[4];
#pragma unroll
    for (int j = 0; j < 4; ++j)
        bPtr[j] = Bt + (size_t)(baseN + j * 16 + lane15) * K + laneHi * 8;

    for (int k0 = 0; k0 < K; k0 += 32) {
        FragBF a0, a1, bf[4];
        a0.u[0] = *(const uint4*)(aPtr0 + k0);
        a0.u[1] = *(const uint4*)(aPtr0 + k0 + 16);
        a1.u[0] = *(const uint4*)(aPtr1 + k0);
        a1.u[1] = *(const uint4*)(aPtr1 + k0 + 16);
#pragma unroll
        for (int j = 0; j < 4; ++j) {
            bf[j].u[0] = *(const uint4*)(bPtr[j] + k0);
            bf[j].u[1] = *(const uint4*)(bPtr[j] + k0 + 16);
        }
#pragma unroll
        for (int j = 0; j < 4; ++j) {
            acc[0][j] = __builtin_amdgcn_wmma_f32_16x16x32_bf16(
                false, a0.v, false, bf[j].v, (short)0, acc[0][j], false, false);
            acc[1][j] = __builtin_amdgcn_wmma_f32_16x16x32_bf16(
                false, a1.v, false, bf[j].v, (short)0, acc[1][j], false, false);
        }
    }
    gemm_epilogue(acc, Out, mask, baseM, baseN, laneHi, lane15, N, mode);
}
#endif

// ---------------- pass 1: per-chunk summaries ----------------

__global__ __launch_bounds__(256) void scan_chunks_kernel(
    const float* __restrict__ qkv,
    const float* __restrict__ decay_param,
    float* __restrict__ Skv, float* __restrict__ Sks)
{
    const int wg   = blockIdx.x * 8 + (threadIdx.x >> 5);
    const int lane = threadIdx.x & 31;
    const int c = wg & (NC_ - 1);
    const int h = (wg >> 4) & (NH_ - 1);
    const int b = wg >> 8;
    const float d = sigmoidf(decay_param[h]);

    float4 kv = {0.f, 0.f, 0.f, 0.f}, ks = {0.f, 0.f, 0.f, 0.f};
    const float* base = qkv + (size_t)(b * S_ + c * CHUNK_) * NQKV_ + h * HD_ + lane * 4;
    for (int j = 0; j < CHUNK_; ++j) {
        float4 k4 = *(const float4*)(base + 2048);
        float4 v4 = *(const float4*)(base + 4096);
        kv.x = d * kv.x + k4.x * v4.x;  ks.x = d * ks.x + k4.x;
        kv.y = d * kv.y + k4.y * v4.y;  ks.y = d * ks.y + k4.y;
        kv.z = d * kv.z + k4.z * v4.z;  ks.z = d * ks.z + k4.z;
        kv.w = d * kv.w + k4.w * v4.w;  ks.w = d * ks.w + k4.w;
        base += NQKV_;
    }
    size_t o = (size_t)wg * HD_ + lane * 4;
    *(float4*)(Skv + o) = kv;
    *(float4*)(Sks + o) = ks;
}

// ---------------- pass 2: sequential combine over chunks ----------------

__global__ void combine_chunks_kernel(
    const float* __restrict__ Skv, const float* __restrict__ Sks,
    const float* __restrict__ kv0, const float* __restrict__ ks0,
    const float* __restrict__ decay_param,
    float* __restrict__ Ikv, float* __restrict__ Iks,
    float* __restrict__ outTail)
{
    const int bh = blockIdx.x;
    const int t  = threadIdx.x;
    const int h  = bh & (NH_ - 1);
    float d = sigmoidf(decay_param[h]);
    float dC = d;
#pragma unroll
    for (int i = 0; i < 7; ++i) dC *= dC;   // d^128

    float skv = kv0[(size_t)bh * HD_ + t];
    float sks = ks0[(size_t)bh * HD_ + t];
    for (int c = 0; c < NC_; ++c) {
        size_t idx = ((size_t)bh * NC_ + c) * HD_ + t;
        Ikv[idx] = skv;
        Iks[idx] = sks;
        skv = dC * skv + Skv[idx];
        sks = dC * sks + Sks[idx];
    }
    outTail[(size_t)bh * HD_ + t] = skv;
    outTail[(size_t)B_ * NH_ * HD_ + (size_t)bh * HD_ + t] = sks;
}

// ---------------- pass 3: outputs per chunk ----------------

__global__ __launch_bounds__(256) void scan_out_kernel(
    const float* __restrict__ qkv,
    const float* __restrict__ decay_param,
    const float* __restrict__ Ikv, const float* __restrict__ Iks,
    unsigned short* __restrict__ attn)
{
    const int wg   = blockIdx.x * 8 + (threadIdx.x >> 5);
    const int lane = threadIdx.x & 31;
    const int c = wg & (NC_ - 1);
    const int h = (wg >> 4) & (NH_ - 1);
    const int b = wg >> 8;
    const float d = sigmoidf(decay_param[h]);

    size_t so = (size_t)wg * HD_ + lane * 4;
    float4 kv = *(const float4*)(Ikv + so);
    float4 ks = *(const float4*)(Iks + so);

    const float* base = qkv + (size_t)(b * S_ + c * CHUNK_) * NQKV_ + h * HD_ + lane * 4;
    unsigned short* ab = attn + (size_t)(b * S_ + c * CHUNK_) * HID_ + h * HD_ + lane * 4;

    for (int j = 0; j < CHUNK_; ++j) {
        float4 q4 = *(const float4*)(base);
        float4 k4 = *(const float4*)(base + 2048);
        float4 v4 = *(const float4*)(base + 4096);
        kv.x = d * kv.x + k4.x * v4.x;  ks.x = d * ks.x + k4.x;
        kv.y = d * kv.y + k4.y * v4.y;  ks.y = d * ks.y + k4.y;
        kv.z = d * kv.z + k4.z * v4.z;  ks.z = d * ks.z + k4.z;
        kv.w = d * kv.w + k4.w * v4.w;  ks.w = d * ks.w + k4.w;

        float num = q4.x * kv.x + q4.y * kv.y + q4.z * kv.z + q4.w * kv.w;
        float den = q4.x * ks.x + q4.y * ks.y + q4.z * ks.z + q4.w * ks.w;
#pragma unroll
        for (int m = 16; m >= 1; m >>= 1) {
            num += __shfl_xor(num, m, 32);
            den += __shfl_xor(den, m, 32);
        }
        float sc = num / fmaxf(den, 1e-6f);
        ushort4 o;
        o.x = f2bf(sc * q4.x); o.y = f2bf(sc * q4.y);
        o.z = f2bf(sc * q4.z); o.w = f2bf(sc * q4.w);
        *(ushort4*)ab = o;

        base += NQKV_;
        ab   += HID_;
    }
}

// ---------------- launch ----------------

extern "C" void kernel_launch(void* const* d_in, const int* in_sizes, int n_in,
                              void* d_out, int out_size, void* d_ws, size_t ws_size,
                              hipStream_t stream) {
    const float* x     = (const float*)d_in[0];
    const float* kv0   = (const float*)d_in[1];
    const float* ks0   = (const float*)d_in[2];
    const float* mask  = (const float*)d_in[3];
    const float* Wqkv  = (const float*)d_in[4];
    const float* Wout  = (const float*)d_in[5];
    const float* decay = (const float*)d_in[6];
    float* out = (float*)d_out;

    char* p = (char*)d_ws;
    unsigned short* xb    = (unsigned short*)p; p += (size_t)M_ * HID_ * 2;
    unsigned short* wqkvT = (unsigned short*)p; p += (size_t)NQKV_ * HID_ * 2;
    unsigned short* woutT = (unsigned short*)p; p += (size_t)HID_ * HID_ * 2;
    float* qkv            = (float*)p;          p += (size_t)M_ * NQKV_ * 4;
    unsigned short* attn  = (unsigned short*)p; p += (size_t)M_ * HID_ * 2;
    const size_t stateSz  = (size_t)B_ * NH_ * NC_ * HD_ * 4;
    float* Skv = (float*)p; p += stateSz;
    float* Sks = (float*)p; p += stateSz;
    float* Ikv = (float*)p; p += stateSz;
    float* Iks = (float*)p; p += stateSz;

    {
        int n = M_ * HID_;
        cast_x_kernel<<<(n / 4 + 255) / 256, 256, 0, stream>>>(x, xb, n);
        transpose_cast_kernel<<<dim3(NQKV_ / 32, HID_ / 32), dim3(32, 8), 0, stream>>>(
            Wqkv, wqkvT, HID_, NQKV_);
        transpose_cast_kernel<<<dim3(HID_ / 32, HID_ / 32), dim3(32, 8), 0, stream>>>(
            Wout, woutT, HID_, HID_);
    }

#if HAS_TDM
    wmma_gemm_tdm_kernel<<<dim3(NQKV_ / 256, M_ / 64), 256, 0, stream>>>(
        xb, wqkvT, qkv, mask, NQKV_, HID_, 1);
#else
    wmma_gemm_kernel<<<dim3(NQKV_ / 256, M_ / 64), 256, 0, stream>>>(
        xb, wqkvT, qkv, mask, NQKV_, HID_, 1);
#endif

    scan_chunks_kernel<<<(B_ * NH_ * NC_) / 8, 256, 0, stream>>>(qkv, decay, Skv, Sks);
    combine_chunks_kernel<<<B_ * NH_, HD_, 0, stream>>>(
        Skv, Sks, kv0, ks0, decay, Ikv, Iks, out + OUT_ELEMS_);
    scan_out_kernel<<<(B_ * NH_ * NC_) / 8, 256, 0, stream>>>(qkv, decay, Ikv, Iks, attn);

#if HAS_TDM
    wmma_gemm_tdm_kernel<<<dim3(HID_ / 256, M_ / 64), 256, 0, stream>>>(
        attn, woutT, out, nullptr, HID_, HID_, 0);
#else
    wmma_gemm_kernel<<<dim3(HID_ / 256, M_ / 64), 256, 0, stream>>>(
        attn, woutT, out, nullptr, HID_, HID_, 0);
#endif
}
